// MonolithicDecoderLayer_66468913873170
// MI455X (gfx1250) — compile-verified, matrix operand order checked by reading, not run on hardware
//
#include <hip/hip_runtime.h>

// ---------------- problem constants ----------------
#define Tt     2048
#define Hh     2048
#define HN     16
#define NOPE   128
#define ROPE   64
#define DV     128
#define QKD    192      // NOPE+ROPE
#define QLORA  768
#define KVLORA 512
#define HIh    16
#define DIh    128
#define TOPK   128
#define NEXP   8
#define IMh    512
#define ISh    512
#define NEGV   (-1.0e30f)
#define RSFv   2.5f
#define LN1E4  9.210340372f   // ln(10000)
#define DI_RS  0.08838834764831845f   // 1/sqrt(128)
#define QK_RS  0.07216878364870323f   // 1/sqrt(192)

typedef __attribute__((ext_vector_type(16))) __bf16 v16bf;
typedef __attribute__((ext_vector_type(8)))  float  v8f;

__device__ inline unsigned short f2bf(float f) {
  union { float f; unsigned u; } c; c.f = f;
  unsigned u = c.u;
  unsigned r = u + 0x7FFFu + ((u >> 16) & 1u);
  return (unsigned short)(r >> 16);
}

__device__ inline v8f wmma_bf16(v16bf a, v16bf b, v8f c) {
  return __builtin_amdgcn_wmma_f32_16x16x32_bf16(false, a, false, b, (short)0, c, false, false);
}

// ---- CDNA5 async global->LDS copy (16B per lane), tracked by ASYNCcnt ----
__device__ inline void async_ld16(const void* g, void* lds) {
  unsigned      l  = (unsigned)(size_t)lds;             // low 32 bits = LDS byte offset
  unsigned long long ga = (unsigned long long)(size_t)g;
  asm volatile("global_load_async_to_lds_b128 %0, %1, off" :: "v"(l), "v"(ga) : "memory");
}
__device__ inline void wait_async() {
  asm volatile("s_wait_asynccnt 0x0" ::: "memory");
}

// A-style 16x32 bf16 fragment from (LDS or global) row-major [row][k] storage.
// lanes 0-15: row = lane, K elems {0..7,16..23}; lanes 16-31: row = lane-16, K {8..15,24..31}.
__device__ inline v16bf ld_frag(const unsigned short* p, int ld, int lane) {
  int r = lane & 15, hf = (lane >> 4) & 1;
  const unsigned short* row = p + (long)r * ld + hf * 8;
  v16bf v;
  unsigned short* tp = (unsigned short*)&v;
#pragma unroll
  for (int e = 0; e < 8; e++) { tp[e] = row[e]; tp[e + 8] = row[16 + e]; }
  return v;
}

// Transposed B fragment: storage is [k][c]; fragment "row" n = lane&15 selects column c0+n,
// elements walk K with the same interleaving as ld_frag.
__device__ inline v16bf ld_frag_t(const unsigned short* p, int ld, int lane) {
  int n = lane & 15, hf = (lane >> 4) & 1;
  v16bf v;
  unsigned short* tp = (unsigned short*)&v;
#pragma unroll
  for (int e = 0; e < 8; e++) {
    tp[e]     = p[(long)(hf * 8 + e) * ld + n];
    tp[e + 8] = p[(long)(16 + hf * 8 + e) * ld + n];
  }
  return v;
}

// ---------------- generic bf16 WMMA GEMM:  C[M,N] = A[M,K] * B[N,K]^T ----------------
// 64x128 block tile, K-step 32. 8 waves: 4 (M) x 2 (N); each wave: 16 rows x 64 cols (4 WMMA).
// Tile staging uses global_load_async_to_lds_b128 (ASYNCcnt) + global_prefetch.
__global__ __launch_bounds__(256) void gemm_bf16_kernel(
    int M, int N, int K,
    const unsigned short* __restrict__ A, int lda,
    const unsigned short* __restrict__ B, int ldb,
    void* __restrict__ Cv, int ldc, int acc, int cbf) {
  __shared__ unsigned short As[64][40];
  __shared__ unsigned short Bs[128][40];
  int tid = threadIdx.x, lane = tid & 31, wave = tid >> 5;
  int mrow = (wave & 3) * 16;
  int ncol = (wave >> 2) * 64;
  int tb = blockIdx.y * 64, nb = blockIdx.x * 128;
  int arow = tid >> 2, acol = (tid & 3) * 8;   // 256 x 16B chunks for A (64x32)
  v8f acc0 = {}, acc1 = {}, acc2 = {}, acc3 = {};
  for (int k0 = 0; k0 < K; k0 += 32) {
    {  // A tile (64x32)
      int gr = tb + arow;
      if (gr < M) {
        async_ld16(A + (long)gr * lda + k0 + acol, &As[arow][acol]);
        if (k0 + 32 < K)
          __builtin_prefetch(A + (long)gr * lda + k0 + 32 + acol, 0, 1);
      } else {
        uint4 z; z.x = z.y = z.z = z.w = 0u;
        *(uint4*)&As[arow][acol] = z;
      }
    }
#pragma unroll
    for (int c = tid; c < 512; c += 256) {  // B tile (128x32): 512 x 16B chunks
      int br = c >> 2, bc = (c & 3) * 8;
      int gb = nb + br;
      if (gb < N) {
        async_ld16(B + (long)gb * ldb + k0 + bc, &Bs[br][bc]);
      } else {
        uint4 z; z.x = z.y = z.z = z.w = 0u;
        *(uint4*)&Bs[br][bc] = z;
      }
    }
    wait_async();
    __syncthreads();
    v16bf af = ld_frag(&As[mrow][0], 40, lane);
    v16bf b0 = ld_frag(&Bs[ncol][0], 40, lane);
    v16bf b1 = ld_frag(&Bs[ncol + 16][0], 40, lane);
    v16bf b2 = ld_frag(&Bs[ncol + 32][0], 40, lane);
    v16bf b3 = ld_frag(&Bs[ncol + 48][0], 40, lane);
    acc0 = wmma_bf16(af, b0, acc0);
    acc1 = wmma_bf16(af, b1, acc1);
    acc2 = wmma_bf16(af, b2, acc2);
    acc3 = wmma_bf16(af, b3, acc3);
    __syncthreads();
  }
  int hf = lane >> 4, cn = lane & 15;
#pragma unroll
  for (int j = 0; j < 8; j++) {
    int r = tb + mrow + hf * 8 + j;
    if (r >= M) continue;
    float vv[4] = {acc0[j], acc1[j], acc2[j], acc3[j]};
#pragma unroll
    for (int b = 0; b < 4; b++) {
      int cc = nb + ncol + b * 16 + cn;
      if (cc >= N) continue;
      if (cbf) {
        ((unsigned short*)Cv)[(long)r * ldc + cc] = f2bf(vv[b]);
      } else {
        float* p = (float*)Cv + (long)r * ldc + cc;
        float v = vv[b];
        if (acc) v += *p;
        *p = v;
      }
    }
  }
}

// ---------------- indexer: logits[t,s] = sum_h iw[t,h]/4 * relu(iq[t,h,:]·ik[s,:] / sqrt(DI)) ----------------
__global__ __launch_bounds__(256) void indexer_kernel(
    const unsigned short* __restrict__ iq,   // (T, HI, DI) bf16
    const unsigned short* __restrict__ ik,   // (T, DI) bf16
    const float* __restrict__ iw,            // (T, HI) f32
    float* __restrict__ logits) {            // (T, T)
  int tb = blockIdx.y * 64, sb = blockIdx.x * 64;
  if (sb > tb + 63) return;  // entirely in causal-masked future; top-k masks it anyway
  __shared__ unsigned short As[64][40];
  __shared__ unsigned short Bs[64][40];
  int tid = threadIdx.x, lane = tid & 31, wave = tid >> 5;
  int mrow = (wave & 3) * 16, ncol = (wave >> 2) * 32;
  int lrow = tid >> 2, lcol = (tid & 3) * 8;
  int hf = lane >> 4, cn = lane & 15;
  float accL0[8] = {0}, accL1[8] = {0};
  for (int h = 0; h < HIh; h++) {
    v8f d0 = {}, d1 = {};
    for (int k0 = 0; k0 < DIh; k0 += 32) {
      async_ld16(iq + ((long)(tb + lrow) * HIh + h) * DIh + k0 + lcol, &As[lrow][lcol]);
      async_ld16(ik + (long)(sb + lrow) * DIh + k0 + lcol, &Bs[lrow][lcol]);
      wait_async();
      __syncthreads();
      v16bf af = ld_frag(&As[mrow][0], 40, lane);
      v16bf b0 = ld_frag(&Bs[ncol][0], 40, lane);
      v16bf b1 = ld_frag(&Bs[ncol + 16][0], 40, lane);
      d0 = wmma_bf16(af, b0, d0);
      d1 = wmma_bf16(af, b1, d1);
      __syncthreads();
    }
#pragma unroll
    for (int j = 0; j < 8; j++) {
      int r = tb + mrow + hf * 8 + j;
      float w = iw[(long)r * HIh + h] * 0.25f;  // HI^-0.5
      accL0[j] += w * fmaxf(d0[j] * DI_RS, 0.f);
      accL1[j] += w * fmaxf(d1[j] * DI_RS, 0.f);
    }
  }
#pragma unroll
  for (int j = 0; j < 8; j++) {
    int r = tb + mrow + hf * 8 + j;
    logits[(long)r * Tt + sb + ncol + cn] = accL0[j];
    logits[(long)r * Tt + sb + ncol + 16 + cn] = accL1[j];
  }
}

// ---------------- top-k(128) with causal mask, per token row ----------------
__global__ __launch_bounds__(256) void topk_kernel(
    const float* __restrict__ logits, int* __restrict__ oidx, float* __restrict__ oval) {
  int t = blockIdx.x, tid = threadIdx.x;
  __shared__ float vals[Tt];
  __shared__ float rv[256];
  __shared__ int   ri[256];
  for (int s = tid; s < Tt; s += 256)
    vals[s] = (s <= t) ? logits[(long)t * Tt + s] : NEGV;
  __syncthreads();
  for (int it = 0; it < TOPK; it++) {
    float bv = -3.0e38f; int bi = 0;
    for (int s = tid; s < Tt; s += 256) {
      float v = vals[s];
      if (v > bv || (v == bv && s < bi)) { bv = v; bi = s; }
    }
    rv[tid] = bv; ri[tid] = bi;
    __syncthreads();
    for (int o = 128; o > 0; o >>= 1) {
      if (tid < o) {
        if (rv[tid + o] > rv[tid] || (rv[tid + o] == rv[tid] && ri[tid + o] < ri[tid])) {
          rv[tid] = rv[tid + o]; ri[tid] = ri[tid + o];
        }
      }
      __syncthreads();
    }
    if (tid == 0) {
      oval[(long)t * TOPK + it] = rv[0];
      oidx[(long)t * TOPK + it] = ri[0];
      vals[ri[0]] = -3.0e38f;
    }
    __syncthreads();
  }
}

// ---------------- per-token sparse MLA attention ----------------
// async-gather 128 selected latent rows -> LDS; scores = (Qabs·KV + Qpe·PE)/sqrt(QK);
// masked softmax; o_lat = attn · KV  (all matmuls via WMMA)
__global__ __launch_bounds__(256) void attn_kernel(
    const unsigned short* __restrict__ qabs,  // (T, HN, 512) bf16
    const unsigned short* __restrict__ qpe,   // (T, HN, 64)  bf16
    const unsigned short* __restrict__ kvc,   // (T, 512) bf16
    const unsigned short* __restrict__ kpe,   // (T, 64)  bf16
    const int* __restrict__ tkidx,            // (T, 128)
    const float* __restrict__ tkval,          // (T, 128)
    unsigned short* __restrict__ olat) {      // (T, HN, 512) bf16
  int t = blockIdx.x, tid = threadIdx.x, lane = tid & 31, wave = tid >> 5;
  __shared__ unsigned short kv[TOPK][520];
  __shared__ unsigned short pe[TOPK][72];
  __shared__ float sc[HN][TOPK];
  __shared__ unsigned short abf[HN][136];
  __shared__ int   sidx[TOPK];
  __shared__ float sval[TOPK];
  if (tid < TOPK) {
    sidx[tid] = tkidx[(long)t * TOPK + tid];
    sval[tid] = tkval[(long)t * TOPK + tid];
  }
  __syncthreads();
  {  // async gather: 2 threads per key row, 16B per op straight into LDS
    int row = tid >> 1, part = tid & 1;
    int src = sidx[row];
    if (src < 0 || src >= Tt) src = 0;
    const unsigned short* s = kvc + (long)src * KVLORA + part * 256;
#pragma unroll
    for (int i = 0; i < 32; i++) async_ld16(s + i * 8, &kv[row][part * 256 + i * 8]);
    if (part == 0) {
      const unsigned short* sp = kpe + (long)src * ROPE;
#pragma unroll
      for (int i = 0; i < 8; i++) async_ld16(sp + i * 8, &pe[row][i * 8]);
    }
  }
  wait_async();
  __syncthreads();
  // ---- scores: M=16 heads, wave covers 16 keys, K = 512 + 64 ----
  v8f s0 = {};
  const unsigned short* qa = qabs + (long)t * HN * KVLORA;
  for (int k0 = 0; k0 < KVLORA; k0 += 32) {
    v16bf a = ld_frag(qa + k0, KVLORA, lane);
    v16bf b = ld_frag(&kv[wave * 16][k0], 520, lane);
    s0 = wmma_bf16(a, b, s0);
  }
  const unsigned short* qp = qpe + (long)t * HN * ROPE;
  for (int k0 = 0; k0 < ROPE; k0 += 32) {
    v16bf a = ld_frag(qp + k0, ROPE, lane);
    v16bf b = ld_frag(&pe[wave * 16][k0], 72, lane);
    s0 = wmma_bf16(a, b, s0);
  }
  int hf = lane >> 4, cn = lane & 15;
  int key = wave * 16 + cn;
  bool vld = (sval[key] > NEGV * 0.5f) && (sidx[key] <= t);
#pragma unroll
  for (int j = 0; j < 8; j++)
    sc[hf * 8 + j][key] = vld ? (s0[j] * QK_RS) : NEGV;
  __syncthreads();
  // ---- softmax per head ----
  if (tid < HN) {
    float mx = -3.0e38f;
    for (int k = 0; k < TOPK; k++) mx = fmaxf(mx, sc[tid][k]);
    float sum = 0.f;
    for (int k = 0; k < TOPK; k++) { float e = __expf(sc[tid][k] - mx); sc[tid][k] = e; sum += e; }
    float inv = 1.0f / sum;
    for (int k = 0; k < TOPK; k++) sc[tid][k] *= inv;
  }
  __syncthreads();
  for (int i = tid; i < HN * TOPK; i += 256)
    abf[i >> 7][i & 127] = f2bf(sc[i >> 7][i & 127]);
  __syncthreads();
  // ---- o_lat: M=16 heads, N=512 (wave covers 64 cols), K=128 keys ----
  int cb = wave * 64;
  v8f o0 = {}, o1 = {}, o2 = {}, o3 = {};
  for (int k0 = 0; k0 < TOPK; k0 += 32) {
    v16bf a = ld_frag(&abf[0][k0], 136, lane);
    v16bf b0 = ld_frag_t(&kv[k0][cb + 0], 520, lane);
    v16bf b1 = ld_frag_t(&kv[k0][cb + 16], 520, lane);
    v16bf b2 = ld_frag_t(&kv[k0][cb + 32], 520, lane);
    v16bf b3 = ld_frag_t(&kv[k0][cb + 48], 520, lane);
    o0 = wmma_bf16(a, b0, o0);
    o1 = wmma_bf16(a, b1, o1);
    o2 = wmma_bf16(a, b2, o2);
    o3 = wmma_bf16(a, b3, o3);
  }
  unsigned short* od = olat + (long)t * HN * KVLORA;
#pragma unroll
  for (int j = 0; j < 8; j++) {
    int hr = hf * 8 + j;
    od[(long)hr * KVLORA + cb + cn]      = f2bf(o0[j]);
    od[(long)hr * KVLORA + cb + 16 + cn] = f2bf(o1[j]);
    od[(long)hr * KVLORA + cb + 32 + cn] = f2bf(o2[j]);
    od[(long)hr * KVLORA + cb + 48 + cn] = f2bf(o3[j]);
  }
}

// ---------------- elementwise / norm / pack kernels ----------------
__global__ __launch_bounds__(256) void conv_bf_kernel(
    const float* __restrict__ s, unsigned short* __restrict__ d, long n) {
  for (long i = (long)blockIdx.x * 256 + threadIdx.x; i < n; i += (long)gridDim.x * 256)
    d[i] = f2bf(s[i]);
}

// dst[b][c][r] = src[b][r][c]  (bf16 convert+transpose)
__global__ __launch_bounds__(256) void transpose_bf_kernel(
    const float* __restrict__ s, unsigned short* __restrict__ d, int B, int R, int C) {
  long n = (long)B * R * C;
  for (long i = (long)blockIdx.x * 256 + threadIdx.x; i < n; i += (long)gridDim.x * 256) {
    long r = i % R; long rem = i / R; long c = rem % C; long b = rem / C;
    d[i] = f2bf(s[(b * R + r) * C + c]);
  }
}

__global__ __launch_bounds__(256) void rmsnorm_bf_kernel(
    const float* __restrict__ x, const float* __restrict__ w,
    unsigned short* __restrict__ out, int Hd) {
  int t = blockIdx.x, tid = threadIdx.x;
  __shared__ float red[256];
  const float* row = x + (long)t * Hd;
  float ss = 0.f;
  for (int i = tid; i < Hd; i += 256) { float v = row[i]; ss += v * v; }
  red[tid] = ss; __syncthreads();
  for (int o = 128; o > 0; o >>= 1) { if (tid < o) red[tid] += red[tid + o]; __syncthreads(); }
  float rs = rsqrtf(red[0] / (float)Hd + 1e-6f);
  for (int i = tid; i < Hd; i += 256) out[(long)t * Hd + i] = f2bf(row[i] * rs * w[i]);
}

// qkv (T,1344): rmsnorm q_c / kv_c, rope k_pe
__global__ __launch_bounds__(256) void pack_qkv_kernel(
    const int* __restrict__ pos, const float* __restrict__ qkv,
    const float* __restrict__ wq, const float* __restrict__ wkv,
    unsigned short* __restrict__ qc, unsigned short* __restrict__ kvc,
    unsigned short* __restrict__ kp) {
  int t = blockIdx.x, tid = threadIdx.x;
  __shared__ float r1[256], r2[256];
  const float* row = qkv + (long)t * (QLORA + KVLORA + ROPE);
  float s1 = 0.f, s2 = 0.f;
  for (int i = tid; i < QLORA; i += 256) { float v = row[i]; s1 += v * v; }
  for (int i = tid; i < KVLORA; i += 256) { float v = row[QLORA + i]; s2 += v * v; }
  r1[tid] = s1; r2[tid] = s2; __syncthreads();
  for (int o = 128; o > 0; o >>= 1) {
    if (tid < o) { r1[tid] += r1[tid + o]; r2[tid] += r2[tid + o]; }
    __syncthreads();
  }
  float rs1 = rsqrtf(r1[0] / (float)QLORA + 1e-6f);
  float rs2 = rsqrtf(r2[0] / (float)KVLORA + 1e-6f);
  for (int i = tid; i < QLORA; i += 256) qc[(long)t * QLORA + i] = f2bf(row[i] * rs1 * wq[i]);
  for (int i = tid; i < KVLORA; i += 256) kvc[(long)t * KVLORA + i] = f2bf(row[QLORA + i] * rs2 * wkv[i]);
  if (tid < 32) {
    float inv = __expf(-(2.f * tid / (float)ROPE) * LN1E4);
    float ang = (float)pos[t] * inv, sn, cs;
    __sincosf(ang, &sn, &cs);
    float x1 = row[QLORA + KVLORA + tid], x2 = row[QLORA + KVLORA + 32 + tid];
    kp[(long)t * ROPE + tid]      = f2bf(x1 * cs - x2 * sn);
    kp[(long)t * ROPE + 32 + tid] = f2bf(x2 * cs + x1 * sn);
  }
}

// index_k: layernorm then rope first 64
__global__ __launch_bounds__(128) void pack_ik_kernel(
    const int* __restrict__ pos, const float* __restrict__ idxk,
    const float* __restrict__ lw, const float* __restrict__ lb,
    unsigned short* __restrict__ ikbf) {
  int t = blockIdx.x, tid = threadIdx.x;
  __shared__ float red[128];
  __shared__ float buf[DIh];
  const float* row = idxk + (long)t * DIh;
  float v = row[tid];
  red[tid] = v; __syncthreads();
  for (int o = 64; o > 0; o >>= 1) { if (tid < o) red[tid] += red[tid + o]; __syncthreads(); }
  float mu = red[0] / (float)DIh; __syncthreads();
  float dv = v - mu;
  red[tid] = dv * dv; __syncthreads();
  for (int o = 64; o > 0; o >>= 1) { if (tid < o) red[tid] += red[tid + o]; __syncthreads(); }
  float rs = rsqrtf(red[0] / (float)DIh + 1e-6f);
  buf[tid] = dv * rs * lw[tid] + lb[tid];
  __syncthreads();
  float outv;
  if (tid < ROPE) {
    int dd = tid & 31;
    float inv = __expf(-(2.f * dd / (float)ROPE) * LN1E4);
    float ang = (float)pos[t] * inv, sn, cs;
    __sincosf(ang, &sn, &cs);
    float x1 = buf[dd], x2 = buf[32 + dd];
    outv = (tid >= 32) ? (x2 * cs + x1 * sn) : (x1 * cs - x2 * sn);
  } else outv = buf[tid];
  ikbf[(long)t * DIh + tid] = f2bf(outv);
}

// q (T,3072) -> qnope bf16 (T,16,128), qpe roped bf16 (T,16,64)
__global__ __launch_bounds__(256) void pack_q_kernel(
    const int* __restrict__ pos, const float* __restrict__ q,
    unsigned short* __restrict__ qn, unsigned short* __restrict__ qp) {
  int t = blockIdx.x, tid = threadIdx.x;
  const float* row = q + (long)t * (HN * QKD);
  for (int i = tid; i < HN * NOPE; i += 256) {
    int h = i >> 7, d = i & 127;
    qn[(long)t * HN * NOPE + i] = f2bf(row[h * QKD + d]);
  }
  for (int i = tid; i < HN * ROPE; i += 256) {
    int h = i >> 6, d = i & 63, dd = d & 31;
    float inv = __expf(-(2.f * dd / (float)ROPE) * LN1E4);
    float ang = (float)pos[t] * inv, sn, cs;
    __sincosf(ang, &sn, &cs);
    float x1 = row[h * QKD + NOPE + dd], x2 = row[h * QKD + NOPE + 32 + dd];
    float v = (d >= 32) ? (x2 * cs + x1 * sn) : (x1 * cs - x2 * sn);
    qp[(long)t * HN * ROPE + i] = f2bf(v);
  }
}

// iq (T,2048) -> rope first 64 of each 128 -> bf16
__global__ __launch_bounds__(256) void pack_iq_kernel(
    const int* __restrict__ pos, const float* __restrict__ iq,
    unsigned short* __restrict__ iqbf) {
  int t = blockIdx.x, tid = threadIdx.x;
  const float* row = iq + (long)t * (HIh * DIh);
  for (int i = tid; i < HIh * DIh; i += 256) {
    int h = i >> 7, d = i & 127;
    float v;
    if (d < ROPE) {
      int dd = d & 31;
      float inv = __expf(-(2.f * dd / (float)ROPE) * LN1E4);
      float ang = (float)pos[t] * inv, sn, cs;
      __sincosf(ang, &sn, &cs);
      float x1 = row[h * DIh + dd], x2 = row[h * DIh + 32 + dd];
      v = (d >= 32) ? (x2 * cs + x1 * sn) : (x1 * cs - x2 * sn);
    } else v = row[h * DIh + d];
    iqbf[(long)t * HIh * DIh + i] = f2bf(v);
  }
}

__global__ __launch_bounds__(256) void moe_gate_kernel(
    const float* __restrict__ glog, float* __restrict__ comb) {
  int t = blockIdx.x * 256 + threadIdx.x;
  if (t >= Tt) return;
  float l[NEXP]; float mx = -3.0e38f;
  for (int e = 0; e < NEXP; e++) { l[e] = glog[(long)t * NEXP + e]; mx = fmaxf(mx, l[e]); }
  float sum = 0.f;
  for (int e = 0; e < NEXP; e++) { l[e] = __expf(l[e] - mx); sum += l[e]; }
  for (int e = 0; e < NEXP; e++) l[e] /= sum;
  int i0 = 0;
  for (int e = 1; e < NEXP; e++) if (l[e] > l[i0]) i0 = e;
  int i1 = (i0 == 0) ? 1 : 0;
  for (int e = 0; e < NEXP; e++) if (e != i0 && l[e] > l[i1]) i1 = e;
  float ws = l[i0] + l[i1];
  for (int e = 0; e < NEXP; e++)
    comb[(long)t * NEXP + e] = (e == i0 || e == i1) ? (l[e] / ws) : 0.f;
}

__global__ __launch_bounds__(256) void act_silu_kernel(
    const float* __restrict__ g, const float* __restrict__ u,
    unsigned short* __restrict__ o, long n) {
  for (long i = (long)blockIdx.x * 256 + threadIdx.x; i < n; i += (long)gridDim.x * 256) {
    float gv = g[i];
    o[i] = f2bf((gv / (1.f + __expf(-gv))) * u[i]);
  }
}

__global__ __launch_bounds__(256) void act_silu_comb_kernel(
    const float* __restrict__ g, const float* __restrict__ u,
    const float* __restrict__ comb, int e, unsigned short* __restrict__ o) {
  long n = (long)Tt * IMh;
  for (long i = (long)blockIdx.x * 256 + threadIdx.x; i < n; i += (long)gridDim.x * 256) {
    int t = (int)(i / IMh);
    float gv = g[i];
    o[i] = f2bf((gv / (1.f + __expf(-gv))) * u[i] * comb[(long)t * NEXP + e] * RSFv);
  }
}

// ---------------- host orchestration ----------------
static void gemm(int M, int N, int K, const void* A, int lda, const void* B, int ldb,
                 void* C, int ldc, int acc, int cbf, hipStream_t st) {
  dim3 g((N + 127) / 128, (M + 63) / 64);
  gemm_bf16_kernel<<<g, 256, 0, st>>>(M, N, K, (const unsigned short*)A, lda,
                                      (const unsigned short*)B, ldb, C, ldc, acc, cbf);
}

extern "C" void kernel_launch(void* const* d_in, const int* in_sizes, int n_in,
                              void* d_out, int out_size, void* d_ws, size_t ws_size,
                              hipStream_t stream) {
  (void)in_sizes; (void)n_in; (void)out_size; (void)ws_size;
  const int*   positions   = (const int*)d_in[0];
  const float* hidden      = (const float*)d_in[1];
  const float* w_in_ln     = (const float*)d_in[2];
  const float* w_qkv_a     = (const float*)d_in[3];
  const float* w_q_a_ln    = (const float*)d_in[4];
  const float* w_kv_a_ln   = (const float*)d_in[5];
  const float* w_idx_k     = (const float*)d_in[6];
  const float* w_idx_k_lnw = (const float*)d_in[7];
  const float* w_idx_k_lnb = (const float*)d_in[8];
  const float* w_idx_w     = (const float*)d_in[9];
  const float* w_q_b       = (const float*)d_in[10];
  const float* w_idx_qb    = (const float*)d_in[11];
  const float* w_uk        = (const float*)d_in[12];
  const float* w_uv        = (const float*)d_in[13];
  const float* w_o         = (const float*)d_in[14];
  const float* w_post_ln   = (const float*)d_in[15];
  const float* gate_w      = (const float*)d_in[16];
  const float* w_gate_e    = (const float*)d_in[17];
  const float* w_up_e      = (const float*)d_in[18];
  const float* w_down_e    = (const float*)d_in[19];
  const float* w_gate_s    = (const float*)d_in[20];
  const float* w_up_s      = (const float*)d_in[21];
  const float* w_down_s    = (const float*)d_in[22];

  unsigned char* base = (unsigned char*)d_ws;
  size_t off = 0;
  auto alloc = [&](size_t bytes) -> void* {
    void* p = base + off;
    off = (off + bytes + 255) & ~(size_t)255;
    return p;
  };
  const long QKVW = (long)(QLORA + KVLORA + ROPE);

  // bf16 weights
  unsigned short* wqkv_bf   = (unsigned short*)alloc(QKVW * Hh * 2);
  unsigned short* widxk_bf  = (unsigned short*)alloc((long)DIh * Hh * 2);
  unsigned short* widxw_bf  = (unsigned short*)alloc((long)HIh * Hh * 2);
  unsigned short* wqb_bf    = (unsigned short*)alloc((long)HN * QKD * QLORA * 2);
  unsigned short* widxqb_bf = (unsigned short*)alloc((long)HIh * DIh * QLORA * 2);
  unsigned short* wukT_bf   = (unsigned short*)alloc((long)HN * KVLORA * NOPE * 2);
  unsigned short* wuvT_bf   = (unsigned short*)alloc((long)HN * DV * KVLORA * 2);
  unsigned short* wo_bf     = (unsigned short*)alloc((long)Hh * HN * DV * 2);
  unsigned short* gatew_bf  = (unsigned short*)alloc((long)NEXP * Hh * 2);
  unsigned short* wge_bf    = (unsigned short*)alloc((long)NEXP * IMh * Hh * 2);
  unsigned short* wue_bf    = (unsigned short*)alloc((long)NEXP * IMh * Hh * 2);
  unsigned short* wde_bf    = (unsigned short*)alloc((long)NEXP * Hh * IMh * 2);
  unsigned short* wgs_bf    = (unsigned short*)alloc((long)ISh * Hh * 2);
  unsigned short* wus_bf    = (unsigned short*)alloc((long)ISh * Hh * 2);
  unsigned short* wds_bf    = (unsigned short*)alloc((long)Hh * ISh * 2);
  // activations
  unsigned short* xn_bf    = (unsigned short*)alloc((long)Tt * Hh * 2);
  float*          qkv_f    = (float*)alloc((long)Tt * QKVW * 4);
  unsigned short* qc_bf    = (unsigned short*)alloc((long)Tt * QLORA * 2);
  unsigned short* kvc_bf   = (unsigned short*)alloc((long)Tt * KVLORA * 2);
  unsigned short* kpe_bf   = (unsigned short*)alloc((long)Tt * ROPE * 2);
  float*          idxk_f   = (float*)alloc((long)Tt * DIh * 4);
  unsigned short* ik_bf    = (unsigned short*)alloc((long)Tt * DIh * 2);
  float*          iw_f     = (float*)alloc((long)Tt * HIh * 4);
  float*          q_f      = (float*)alloc((long)Tt * HN * QKD * 4);
  unsigned short* qnope_bf = (unsigned short*)alloc((long)Tt * HN * NOPE * 2);
  unsigned short* qpe_bf   = (unsigned short*)alloc((long)Tt * HN * ROPE * 2);
  float*          iq_f     = (float*)alloc((long)Tt * HIh * DIh * 4);
  unsigned short* iq_bf    = (unsigned short*)alloc((long)Tt * HIh * DIh * 2);
  float*          logits   = (float*)alloc((long)Tt * Tt * 4);
  int*            tk_idx   = (int*)alloc((long)Tt * TOPK * 4);
  float*          tk_val   = (float*)alloc((long)Tt * TOPK * 4);
  unsigned short* qabs_bf  = (unsigned short*)alloc((long)Tt * HN * KVLORA * 2);
  unsigned short* olat_bf  = (unsigned short*)alloc((long)Tt * HN * KVLORA * 2);
  unsigned short* o_bf     = (unsigned short*)alloc((long)Tt * HN * DV * 2);
  unsigned short* x2_bf    = (unsigned short*)alloc((long)Tt * Hh * 2);
  float*          glog_f   = (float*)alloc((long)Tt * NEXP * 4);
  float*          comb_f   = (float*)alloc((long)Tt * NEXP * 4);
  float*          gbuf     = (float*)alloc((long)Tt * IMh * 4);
  float*          ubuf     = (float*)alloc((long)Tt * IMh * 4);
  unsigned short* act_bf   = (unsigned short*)alloc((long)Tt * IMh * 2);

  float* outp = (float*)d_out;
  float* res2 = (float*)d_out + (long)Tt * Hh;

  auto conv = [&](const float* s, unsigned short* d, long n) {
    long blocks = (n + 255) / 256; if (blocks > 8192) blocks = 8192;
    conv_bf_kernel<<<(int)blocks, 256, 0, stream>>>(s, d, n);
  };
  // weight conversions (bf16)
  conv(w_qkv_a, wqkv_bf, QKVW * Hh);
  conv(w_idx_k, widxk_bf, (long)DIh * Hh);
  conv(w_idx_w, widxw_bf, (long)HIh * Hh);
  conv(w_q_b, wqb_bf, (long)HN * QKD * QLORA);
  conv(w_idx_qb, widxqb_bf, (long)HIh * DIh * QLORA);
  conv(w_o, wo_bf, (long)Hh * HN * DV);
  conv(gate_w, gatew_bf, (long)NEXP * Hh);
  conv(w_gate_e, wge_bf, (long)NEXP * IMh * Hh);
  conv(w_up_e, wue_bf, (long)NEXP * IMh * Hh);
  conv(w_down_e, wde_bf, (long)NEXP * Hh * IMh);
  conv(w_gate_s, wgs_bf, (long)ISh * Hh);
  conv(w_up_s, wus_bf, (long)ISh * Hh);
  conv(w_down_s, wds_bf, (long)Hh * ISh);
  transpose_bf_kernel<<<4096, 256, 0, stream>>>(w_uk, wukT_bf, HN, NOPE, KVLORA);  // -> (h,512,128)
  transpose_bf_kernel<<<4096, 256, 0, stream>>>(w_uv, wuvT_bf, HN, KVLORA, DV);    // -> (h,128,512)

  // 1) input RMSNorm -> bf16
  rmsnorm_bf_kernel<<<Tt, 256, 0, stream>>>(hidden, w_in_ln, xn_bf, Hh);
  // 2) low-rank projections
  gemm(Tt, (int)QKVW, Hh, xn_bf, Hh, wqkv_bf, Hh, qkv_f, (int)QKVW, 0, 0, stream);
  gemm(Tt, DIh, Hh, xn_bf, Hh, widxk_bf, Hh, idxk_f, DIh, 0, 0, stream);
  gemm(Tt, HIh, Hh, xn_bf, Hh, widxw_bf, Hh, iw_f, HIh, 0, 0, stream);
  // 3) split/norm/rope
  pack_qkv_kernel<<<Tt, 256, 0, stream>>>(positions, qkv_f, w_q_a_ln, w_kv_a_ln,
                                          qc_bf, kvc_bf, kpe_bf);
  pack_ik_kernel<<<Tt, 128, 0, stream>>>(positions, idxk_f, w_idx_k_lnw, w_idx_k_lnb, ik_bf);
  // 4) q_b / idx_qb projections
  gemm(Tt, HN * QKD, QLORA, qc_bf, QLORA, wqb_bf, QLORA, q_f, HN * QKD, 0, 0, stream);
  pack_q_kernel<<<Tt, 256, 0, stream>>>(positions, q_f, qnope_bf, qpe_bf);
  gemm(Tt, HIh * DIh, QLORA, qc_bf, QLORA, widxqb_bf, QLORA, iq_f, HIh * DIh, 0, 0, stream);
  pack_iq_kernel<<<Tt, 256, 0, stream>>>(positions, iq_f, iq_bf);
  // 5) indexer logits + causal top-k
  indexer_kernel<<<dim3(Tt / 64, Tt / 64), 256, 0, stream>>>(iq_bf, ik_bf, iw_f, logits);
  topk_kernel<<<Tt, 256, 0, stream>>>(logits, tk_idx, tk_val);
  // 6) q_abs = q_nope @ w_uk (per head), bf16 out
  for (int h = 0; h < HN; h++)
    gemm(Tt, KVLORA, NOPE, qnope_bf + h * NOPE, HN * NOPE,
         wukT_bf + (long)h * KVLORA * NOPE, NOPE,
         qabs_bf + h * KVLORA, HN * KVLORA, 0, 1, stream);
  // 7) sparse attention
  attn_kernel<<<Tt, 256, 0, stream>>>(qabs_bf, qpe_bf, kvc_bf, kpe_bf, tk_idx, tk_val, olat_bf);
  // 8) o = o_lat @ w_uv (per head), bf16 out
  for (int h = 0; h < HN; h++)
    gemm(Tt, DV, KVLORA, olat_bf + h * KVLORA, HN * KVLORA,
         wuvT_bf + (long)h * DV * KVLORA, KVLORA,
         o_bf + h * DV, HN * DV, 0, 1, stream);
  // 9) output proj + residual (residual2 = second half of d_out)
  hipMemcpyAsync(res2, hidden, (size_t)Tt * Hh * 4, hipMemcpyDeviceToDevice, stream);
  gemm(Tt, Hh, HN * DV, o_bf, HN * DV, wo_bf, HN * DV, res2, Hh, 1, 0, stream);
  // 10) post-LN + MoE gate
  rmsnorm_bf_kernel<<<Tt, 256, 0, stream>>>(res2, w_post_ln, x2_bf, Hh);
  gemm(Tt, NEXP, Hh, x2_bf, Hh, gatew_bf, Hh, glog_f, NEXP, 0, 0, stream);
  moe_gate_kernel<<<(Tt + 255) / 256, 256, 0, stream>>>(glog_f, comb_f);
  // 11) shared expert -> out (acc=0)
  gemm(Tt, ISh, Hh, x2_bf, Hh, wgs_bf, Hh, gbuf, ISh, 0, 0, stream);
  gemm(Tt, ISh, Hh, x2_bf, Hh, wus_bf, Hh, ubuf, ISh, 0, 0, stream);
  act_silu_kernel<<<4096, 256, 0, stream>>>(gbuf, ubuf, act_bf, (long)Tt * ISh);
  gemm(Tt, Hh, ISh, act_bf, ISh, wds_bf, ISh, outp, Hh, 0, 0, stream);
  // 12) routed experts (dense, weighted by comb*RSF) accumulate into out
  for (int e = 0; e < NEXP; e++) {
    gemm(Tt, IMh, Hh, x2_bf, Hh, wge_bf + (long)e * IMh * Hh, Hh, gbuf, IMh, 0, 0, stream);
    gemm(Tt, IMh, Hh, x2_bf, Hh, wue_bf + (long)e * IMh * Hh, Hh, ubuf, IMh, 0, 0, stream);
    act_silu_comb_kernel<<<4096, 256, 0, stream>>>(gbuf, ubuf, comb_f, e, act_bf);
    gemm(Tt, Hh, IMh, act_bf, IMh, wde_bf + (long)e * Hh * IMh, IMh, outp, Hh, 1, 0, stream);
  }
}
